// TrajectoryTransformer_26809185861664
// MI455X (gfx1250) — matrix operand order repaired and checked by reading, not compiled, not run on hardware
//
#include <hip/hip_runtime.h>

typedef __attribute__((ext_vector_type(16))) _Float16 v16h_t;
typedef __attribute__((ext_vector_type(8)))  _Float16 v8h_t;
typedef __attribute__((ext_vector_type(8)))  float    v8f_t;

// ---------------- f16 weight workspace layout (offsets in halves) ----------------
constexpr int OFF_AIN  = 0;                  // attn_in_w  2 x 192 x 64
constexpr int OFF_AOUT = 24576;              // attn_out_w 2 x 64 x 64
constexpr int OFF_FF1  = 32768;              // ff1_w      2 x 256 x 64
constexpr int OFF_FF2  = 65536;              // ff2_w      2 x 64 x 256
constexpr int OFF_SIN  = 98304;              // soc_in_w   192 x 64
constexpr int OFF_G1   = 110592;             // goal1_w    64 x 128
constexpr int OFF_T1   = 118784;             // traj1_w padded 128 x 136 (cols>=130 zero)
constexpr int OFF_T2   = 136192;             // traj2_w padded 32 x 128 (rows>=24 zero)
constexpr int WF16_TOT = 140288;

// ---------------- ws byte offsets ----------------
constexpr size_t WS_H32  = 393216;                       // h (ego) f32, 8192*64
constexpr size_t WS_NH16 = 2490368;                      // neighbor h f16, 8192*16*64
constexpr size_t WS_HF   = 19267584;                     // h_final f32, 8192*128

// ---------------- d_out float offsets ----------------
constexpr int OUT_TRAJ  = 0;        // (B,3,12,2) = 589824
constexpr int OUT_GOALS = 589824;   // (B,3,2)    = 49152
constexpr int OUT_PROBS = 638976;   // (B,3)      = 24576
constexpr int OUT_ATTN  = 663552;   // (B,1,16)   = 131072

// ---------------- WMMA helpers ----------------
__device__ __forceinline__ v16h_t make_v16(v8h_t lo, v8h_t hi) {
  v16h_t r;
#pragma unroll
  for (int i = 0; i < 8; ++i) { r[i] = lo[i]; r[i + 8] = hi[i]; }
  return r;
}
// A fragment (16x32 f16, M=lane&15): K chunks at klo and 16+klo (8 halves each)
__device__ __forceinline__ v16h_t fragA(const _Float16* X, int stride, int k0) {
  int lane = threadIdx.x & 31;
  const _Float16* p = X + (lane & 15) * stride + k0 + ((lane >> 4) << 3);
  return make_v16(*(const v8h_t*)p, *(const v8h_t*)(p + 16));
}
// B fragment (32x16 f16, N=lane&15): K chunks at klo and 16+klo (8 halves each)
__device__ __forceinline__ v16h_t fragB(const _Float16* W, int ldk, int n0, int k0) {
  int lane = threadIdx.x & 31;
  const _Float16* p = W + (n0 + (lane & 15)) * ldk + k0 + ((lane >> 4) << 3);
  return make_v16(*(const v8h_t*)p, *(const v8h_t*)(p + 16));
}
__device__ __forceinline__ v8f_t wmma_f16(v16h_t a, v16h_t b, v8f_t c) {
  return __builtin_amdgcn_wmma_f32_16x16x32_f16(false, a, false, b, (short)0, c, false, false);
}
__device__ __forceinline__ int c_col()  { return threadIdx.x & 15; }
__device__ __forceinline__ int c_row0() { return (threadIdx.x & 16) ? 8 : 0; }

// block-wide copy of a weight panel (global f16 -> LDS), count multiple of 8 halves
__device__ __forceinline__ void stage_w(const _Float16* __restrict__ src,
                                        _Float16* __restrict__ dst, int halves) {
  for (int i = threadIdx.x * 8; i < halves; i += blockDim.x * 8)
    *(v8h_t*)(dst + i) = *(const v8h_t*)(src + i);
}

// ---------------- kernel 0: weight conversion f32 -> f16 ----------------
__global__ void prep_kernel(const float* __restrict__ ain, const float* __restrict__ aout,
                            const float* __restrict__ ff1, const float* __restrict__ ff2,
                            const float* __restrict__ sin_, const float* __restrict__ g1,
                            const float* __restrict__ t1, const float* __restrict__ t2,
                            _Float16* __restrict__ w) {
  int i = blockIdx.x * blockDim.x + threadIdx.x;
  if (i >= WF16_TOT) return;
  float v;
  if      (i < OFF_AOUT) v = ain [i - OFF_AIN];
  else if (i < OFF_FF1)  v = aout[i - OFF_AOUT];
  else if (i < OFF_FF2)  v = ff1 [i - OFF_FF1];
  else if (i < OFF_SIN)  v = ff2 [i - OFF_FF2];
  else if (i < OFF_G1)   v = sin_[i - OFF_SIN];
  else if (i < OFF_T1)   v = g1  [i - OFF_G1];
  else if (i < OFF_T2) { int j = i - OFF_T1; int n = j / 136, k = j % 136; v = (k < 130) ? t1[n * 130 + k] : 0.f; }
  else                 { int j = i - OFF_T2; int n = j >> 7,  k = j & 127; v = (n < 24)  ? t2[n * 128 + k] : 0.f; }
  w[i] = (_Float16)v;
}

// ---------------- per-item attention core (VALU) ----------------
__device__ __forceinline__ void attn_item(const _Float16* QKV, int i, float o[16]) {
  int s = i >> 4, h = (i >> 2) & 3, q = i & 3, qr = s * 4 + q;
  float scv[4]; float mx = -1e30f;
#pragma unroll
  for (int k = 0; k < 4; ++k) {
    float acc = 0.f; int kr = s * 4 + k;
#pragma unroll
    for (int d = 0; d < 16; ++d)
      acc += (float)QKV[qr * 208 + h * 16 + d] * (float)QKV[kr * 208 + 64 + h * 16 + d];
    scv[k] = acc * 0.25f; mx = fmaxf(mx, scv[k]);
  }
  float su = 0.f;
#pragma unroll
  for (int k = 0; k < 4; ++k) { scv[k] = __expf(scv[k] - mx); su += scv[k]; }
  float inv = 1.f / su;
#pragma unroll
  for (int d = 0; d < 16; ++d) {
    float acc = 0.f;
#pragma unroll
    for (int k = 0; k < 4; ++k) acc += scv[k] * (float)QKV[(s * 4 + k) * 208 + 128 + h * 16 + d];
    o[d] = acc * inv;
  }
}

__device__ __forceinline__ void ln_row(const float* X32, _Float16* Xh,
                                       const float* g, const float* bb, int lane) {
  if (lane < 16) {
    const float* r = X32 + lane * 64;
    float m = 0.f;
    for (int d = 0; d < 64; ++d) m += r[d];
    m *= (1.0f / 64.0f);
    float v = 0.f;
    for (int d = 0; d < 64; ++d) { float t = r[d] - m; v += t * t; }
    v *= (1.0f / 64.0f);
    float inv = rsqrtf(v + 1e-5f);
    for (int d = 0; d < 64; ++d)
      Xh[lane * 72 + d] = (_Float16)((r[d] - m) * inv * g[d] + bb[d]);
  }
}

// ---------------- kernel 1: fused 2-layer encoder ----------------
// 256 threads = 8 waves, one wave = 4 sequences (16 token rows).
// LDS: 8 x 14848 B per-wave buffers + 32 KB staged weight panel = 151552 B.
constexpr int ENC_WAVE_B   = 14848;
constexpr int ENC_WB_OFF   = 8 * ENC_WAVE_B;           // 118784
constexpr int ENC_LDS_TOT  = ENC_WB_OFF + 32768;       // 151552
__global__ void __launch_bounds__(256) enc_kernel(
    const float* __restrict__ x, const float* __restrict__ nb, const float* __restrict__ pe,
    const float* __restrict__ ew, const float* __restrict__ eb,
    const float* __restrict__ ainb, const float* __restrict__ aoutb,
    const float* __restrict__ f1b, const float* __restrict__ f2b,
    const float* __restrict__ l1g, const float* __restrict__ l1b,
    const float* __restrict__ l2g, const float* __restrict__ l2b,
    const _Float16* __restrict__ wf16,
    float* __restrict__ H32, _Float16* __restrict__ NH16, float* __restrict__ HF) {
  extern __shared__ char smem[];
  const int lane = threadIdx.x & 31;
  const int wave = threadIdx.x >> 5;
  char* wl = smem + wave * ENC_WAVE_B;
  _Float16* Xh  = (_Float16*)wl;            // [16][72]  f16 activation
  float*    X32 = (float*)(wl + 2304);      // [16][64]  f32 scratch (residual/LN)
  char*     SCR = wl + 6400;                // 8448 B union: QKV [16][208] f16 / H1 [16][264] f16
  _Float16* QKV = (_Float16*)SCR;
  _Float16* H1  = (_Float16*)SCR;
  _Float16* WB  = (_Float16*)(smem + ENC_WB_OFF);  // staged weight panel (block-shared)

  // warm L2 with the first projection's weights (global_prefetch_b8 path)
  for (int i = threadIdx.x; i < 384; i += blockDim.x)
    __builtin_prefetch((const char*)wf16 + i * 64, 0, 1);

  const int seqBase = (blockIdx.x * 8 + wave) * 4;

  // stage tokens (reuse X32 as float scratch)
  float* tokf = X32;
  for (int i = lane; i < 112; i += 32) {
    int s = i / 28, r = i % 28, t = r / 7, f = r % 7;
    int seq = seqBase + s;
    tokf[i] = (seq < 8192) ? x[(seq * 4 + t) * 7 + f] : nb[((seq - 8192) * 4 + t) * 7 + f];
  }
  // embed + positional encoding -> Xh (f16)
  for (int r = 0; r < 16; ++r) {
    int t = r & 3;
    float a0 = eb[lane]      + pe[t * 64 + lane];
    float a1 = eb[lane + 32] + pe[t * 64 + lane + 32];
#pragma unroll
    for (int f = 0; f < 7; ++f) {
      float tv = tokf[(r >> 2) * 28 + t * 7 + f];
      a0 += tv * ew[lane * 7 + f];
      a1 += tv * ew[(lane + 32) * 7 + f];
    }
    Xh[r * 72 + lane]      = (_Float16)a0;
    Xh[r * 72 + lane + 32] = (_Float16)a1;
  }

  for (int l = 0; l < 2; ++l) {
    // ---- stage + QKV projection: 12 N-tiles x 2 K-steps (WMMA, B from LDS) ----
    __syncthreads();
    stage_w(wf16 + OFF_AIN + l * (192 * 64), WB, 192 * 64);
    __syncthreads();
    const float* ba = ainb + l * 192;
    for (int nt = 0; nt < 12; ++nt) {
      v8f_t c = {};
      for (int k0 = 0; k0 < 64; k0 += 32)
        c = wmma_f16(fragA(Xh, 72, k0), fragB(WB, 64, nt * 16, k0), c);
      int col = nt * 16 + c_col(); int r0 = c_row0();
      float bv = ba[col];
#pragma unroll
      for (int j = 0; j < 8; ++j) QKV[(r0 + j) * 208 + col] = (_Float16)(c[j] + bv);
    }
    // ---- attention core (block-diagonal 4x4 per head, VALU, per-wave) ----
    float o0[16], o1[16];
    attn_item(QKV, lane, o0);
    attn_item(QKV, lane + 32, o1);
#pragma unroll
    for (int ii = 0; ii < 2; ++ii) {
      int i = lane + 32 * ii;
      int s = i >> 4, h = (i >> 2) & 3, q = i & 3, qr = s * 4 + q;
#pragma unroll
      for (int d = 0; d < 16; ++d)
        QKV[qr * 208 + h * 16 + d] = (_Float16)(ii ? o1[d] : o0[d]);
    }
    // ---- stage + out projection + residual (WMMA) ----
    __syncthreads();
    stage_w(wf16 + OFF_AOUT + l * 4096, WB, 4096);
    __syncthreads();
    for (int nt = 0; nt < 4; ++nt) {
      v8f_t c = {};
      for (int k0 = 0; k0 < 64; k0 += 32)
        c = wmma_f16(fragA(QKV, 208, k0), fragB(WB, 64, nt * 16, k0), c);
      int col = nt * 16 + c_col(); int r0 = c_row0();
      float bv = aoutb[l * 64 + col];
#pragma unroll
      for (int j = 0; j < 8; ++j)
        X32[(r0 + j) * 64 + col] = c[j] + bv + (float)Xh[(r0 + j) * 72 + col];
    }
    ln_row(X32, Xh, l1g + l * 64, l1b + l * 64, lane);
    // ---- stage + FF1: 16 N-tiles (WMMA) + ReLU ----
    __syncthreads();
    stage_w(wf16 + OFF_FF1 + l * (256 * 64), WB, 256 * 64);
    __syncthreads();
    for (int nt = 0; nt < 16; ++nt) {
      v8f_t c = {};
      for (int k0 = 0; k0 < 64; k0 += 32)
        c = wmma_f16(fragA(Xh, 72, k0), fragB(WB, 64, nt * 16, k0), c);
      int col = nt * 16 + c_col(); int r0 = c_row0();
      float bv = f1b[l * 256 + col];
#pragma unroll
      for (int j = 0; j < 8; ++j)
        H1[(r0 + j) * 264 + col] = (_Float16)fmaxf(c[j] + bv, 0.f);
    }
    // ---- stage + FF2: K=256 (WMMA) + residual ----
    __syncthreads();
    stage_w(wf16 + OFF_FF2 + l * (64 * 256), WB, 64 * 256);
    __syncthreads();
    for (int nt = 0; nt < 4; ++nt) {
      v8f_t c = {};
      for (int k0 = 0; k0 < 256; k0 += 32)
        c = wmma_f16(fragA(H1, 264, k0), fragB(WB, 256, nt * 16, k0), c);
      int col = nt * 16 + c_col(); int r0 = c_row0();
      float bv = f2b[l * 64 + col];
#pragma unroll
      for (int j = 0; j < 8; ++j)
        X32[(r0 + j) * 64 + col] = c[j] + bv + (float)Xh[(r0 + j) * 72 + col];
    }
    ln_row(X32, Xh, l2g + l * 64, l2b + l * 64, lane);
  }
  // ---- emit last-token hidden state ----
  for (int i = lane; i < 256; i += 32) {
    int s = i >> 6, d = i & 63;
    int seq = seqBase + s;
    float v = (float)Xh[(s * 4 + 3) * 72 + d];
    if (seq < 8192) { H32[seq * 64 + d] = v; HF[seq * 128 + d] = v; }
    else            { NH16[(size_t)(seq - 8192) * 64 + d] = (_Float16)v; }
  }
}

// ---------------- kernel 2: social attention (1 query x 16 neighbors) ----------------
constexpr int SOC_LDSW = 11584;
__global__ void __launch_bounds__(128) soc_kernel(
    const _Float16* __restrict__ wf16, const float* __restrict__ H32,
    const _Float16* __restrict__ NH16, const float* __restrict__ sinw,
    const float* __restrict__ sinb, const float* __restrict__ soutw,
    const float* __restrict__ soutb, float* __restrict__ HF, float* __restrict__ out) {
  extern __shared__ char smem[];
  const int lane = threadIdx.x & 31;
  const int wave = threadIdx.x >> 5;
  char* wl = smem + wave * SOC_LDSW;
  _Float16* A  = (_Float16*)wl;            // [16][72] neighbor hiddens f16
  float* Kp  = (float*)(wl + 2304);        // [16][64]
  float* Vp  = (float*)(wl + 6400);        // [16][64]
  float* qv  = (float*)(wl + 10496);       // [64]
  float* sc  = (float*)(wl + 10752);       // [64]
  float* hs  = (float*)(wl + 11008);       // [8] max/invsum per head
  float* aw  = (float*)(wl + 11072);       // [64]
  float* o64 = (float*)(wl + 11328);       // [64]
  const int b = blockIdx.x * 4 + wave;

  const _Float16* src = NH16 + (size_t)b * 1024;
  for (int i = lane; i < 128; i += 32) {
    int r = i >> 3, c8 = (i & 7) * 8;
    *(v8h_t*)(A + r * 72 + c8) = *(const v8h_t*)(src + r * 64 + c8);
  }
  // K,V projections via WMMA (M = 16 neighbors)
  const _Float16* Wk = wf16 + OFF_SIN + 64 * 64;
  const _Float16* Wv = wf16 + OFF_SIN + 128 * 64;
  for (int nt = 0; nt < 4; ++nt) {
    v8f_t ck = {}, cv = {};
    for (int k0 = 0; k0 < 64; k0 += 32) {
      v16h_t a = fragA(A, 72, k0);
      ck = wmma_f16(a, fragB(Wk, 64, nt * 16, k0), ck);
      cv = wmma_f16(a, fragB(Wv, 64, nt * 16, k0), cv);
    }
    int col = nt * 16 + c_col(); int r0 = c_row0();
    float bk = sinb[64 + col], bv = sinb[128 + col];
#pragma unroll
    for (int j = 0; j < 8; ++j) {
      Kp[(r0 + j) * 64 + col] = ck[j] + bk;
      Vp[(r0 + j) * 64 + col] = cv[j] + bv;
    }
  }
  // query projection (1 token, VALU)
  for (int ii = 0; ii < 2; ++ii) {
    int d = lane + 32 * ii;
    float acc = sinb[d];
    const float* hrow = H32 + b * 64;
    for (int j = 0; j < 64; ++j) acc += hrow[j] * sinw[d * 64 + j];
    qv[d] = acc;
  }
  // scores, softmax over 16 keys per head
  for (int ii = 0; ii < 2; ++ii) {
    int i = lane + 32 * ii; int h = i >> 4, n = i & 15;
    float s = 0.f;
#pragma unroll
    for (int d = 0; d < 16; ++d) s += qv[h * 16 + d] * Kp[n * 64 + h * 16 + d];
    sc[i] = s * 0.25f;
  }
  if (lane < 4) {
    float m = -1e30f;
    for (int n = 0; n < 16; ++n) m = fmaxf(m, sc[lane * 16 + n]);
    float su = 0.f;
    for (int n = 0; n < 16; ++n) su += __expf(sc[lane * 16 + n] - m);
    hs[lane] = m; hs[4 + lane] = 1.f / su;
  }
  for (int ii = 0; ii < 2; ++ii) {
    int i = lane + 32 * ii; int h = i >> 4;
    aw[i] = __expf(sc[i] - hs[h]) * hs[4 + h];
  }
  if (lane < 16) {  // attn_w = mean over heads
    float s = 0.f;
    for (int h = 0; h < 4; ++h) s += aw[h * 16 + lane];
    out[OUT_ATTN + b * 16 + lane] = s * 0.25f;
  }
  for (int ii = 0; ii < 2; ++ii) {
    int d = lane + 32 * ii; int h = d >> 4;
    float acc = 0.f;
#pragma unroll
    for (int n = 0; n < 16; ++n) acc += aw[h * 16 + n] * Vp[n * 64 + d];
    o64[d] = acc;
  }
  for (int ii = 0; ii < 2; ++ii) {  // output projection, write h_final upper half
    int d = lane + 32 * ii;
    float acc = soutb[d];
    for (int j = 0; j < 64; ++j) acc += o64[j] * soutw[d * 64 + j];
    HF[b * 128 + 64 + d] = acc;
  }
}

// ---------------- kernel 3: prediction heads ----------------
constexpr int HEADS_LDSW = 19712;
__global__ void __launch_bounds__(128) heads_kernel(
    const _Float16* __restrict__ wf16, const float* __restrict__ HF,
    const float* __restrict__ g1b, const float* __restrict__ g2w, const float* __restrict__ g2b,
    const float* __restrict__ t1w, const float* __restrict__ t1b,
    const float* __restrict__ t2b, const float* __restrict__ pw, const float* __restrict__ pb,
    float* __restrict__ out) {
  extern __shared__ char smem[];
  const int lane = threadIdx.x & 31;
  const int wave = threadIdx.x >> 5;
  char* wl = smem + wave * HEADS_LDSW;
  _Float16* Hf   = (_Float16*)wl;            // [16][136] h_final f16
  _Float16* G1   = (_Float16*)(wl + 4352);   // [16][72]
  float*    base = (float*)(wl + 6656);      // [16][128]
  _Float16* T1   = (_Float16*)(wl + 14848);  // [16][136]
  float*    gl   = (float*)(wl + 19200);     // [16][8]
  const int rowBase = (blockIdx.x * 4 + wave) * 16;

  for (int i = lane; i < 16 * 136; i += 32) {
    int r = i / 136, c = i % 136;
    Hf[i] = (c < 128) ? (_Float16)HF[(rowBase + r) * 128 + c] : (_Float16)0.f;
  }
  // goal hidden: K=128 WMMA + ReLU
  const _Float16* Wg = wf16 + OFF_G1;
  for (int nt = 0; nt < 4; ++nt) {
    v8f_t c = {};
    for (int k0 = 0; k0 < 128; k0 += 32)
      c = wmma_f16(fragA(Hf, 136, k0), fragB(Wg, 128, nt * 16, k0), c);
    int col = nt * 16 + c_col(); int r0 = c_row0();
    float bv = g1b[col];
#pragma unroll
    for (int j = 0; j < 8; ++j) G1[(r0 + j) * 72 + col] = (_Float16)fmaxf(c[j] + bv, 0.f);
  }
  // goals (6 outputs, VALU)
  for (int it = 0; it < 3; ++it) {
    int i = lane + 32 * it; int r = i / 6, o = i % 6;
    float acc = g2b[o];
    for (int j = 0; j < 64; ++j) acc += (float)G1[r * 72 + j] * g2w[o * 64 + j];
    gl[r * 8 + o] = acc;
    out[OUT_GOALS + (rowBase + r) * 6 + o] = acc;
  }
  // traj base = hf @ t1w[:, :128]^T (shared across modes, WMMA)
  const _Float16* Wt1 = wf16 + OFF_T1;
  for (int nt = 0; nt < 8; ++nt) {
    v8f_t c = {};
    for (int k0 = 0; k0 < 128; k0 += 32)
      c = wmma_f16(fragA(Hf, 136, k0), fragB(Wt1, 136, nt * 16, k0), c);
    int col = nt * 16 + c_col(); int r0 = c_row0();
    float bv = t1b[col];
#pragma unroll
    for (int j = 0; j < 8; ++j) base[(r0 + j) * 128 + col] = c[j] + bv;
  }
  for (int i = lane; i < 128; i += 32) { int r = i >> 3, c = 128 + (i & 7); T1[r * 136 + c] = (_Float16)0.f; }
  const _Float16* Wt2 = wf16 + OFF_T2;
  for (int mode = 0; mode < 3; ++mode) {
    for (int i = lane; i < 2048; i += 32) {
      int r = i >> 7, n = i & 127;
      float t = base[r * 128 + n] + gl[r * 8 + mode * 2] * t1w[n * 130 + 128]
                                  + gl[r * 8 + mode * 2 + 1] * t1w[n * 130 + 129];
      T1[r * 136 + n] = (_Float16)fmaxf(t, 0.f);
    }
    for (int nt = 0; nt < 2; ++nt) {
      v8f_t c = {};
      for (int k0 = 0; k0 < 128; k0 += 32)
        c = wmma_f16(fragA(T1, 136, k0), fragB(Wt2, 128, nt * 16, k0), c);
      int col = nt * 16 + c_col(); int r0 = c_row0();
      if (col < 24) {
        float bv = t2b[col];
#pragma unroll
        for (int j = 0; j < 8; ++j)
          out[OUT_TRAJ + (rowBase + r0 + j) * 72 + mode * 24 + col] = c[j] + bv;
      }
    }
  }
  // probabilities (3-way softmax, VALU)
  if (lane < 16) {
    float pv[3];
    for (int o = 0; o < 3; ++o) {
      float acc = pb[o];
      for (int j = 0; j < 128; ++j) acc += (float)Hf[lane * 136 + j] * pw[o * 128 + j];
      pv[o] = acc;
    }
    float m = fmaxf(pv[0], fmaxf(pv[1], pv[2]));
    float e0 = __expf(pv[0] - m), e1 = __expf(pv[1] - m), e2 = __expf(pv[2] - m);
    float s = 1.f / (e0 + e1 + e2);
    int gr = rowBase + lane;
    out[OUT_PROBS + gr * 3 + 0] = e0 * s;
    out[OUT_PROBS + gr * 3 + 1] = e1 * s;
    out[OUT_PROBS + gr * 3 + 2] = e2 * s;
  }
}

extern "C" void kernel_launch(void* const* d_in, const int* in_sizes, int n_in,
                              void* d_out, int out_size, void* d_ws, size_t ws_size,
                              hipStream_t stream) {
  (void)in_sizes; (void)n_in; (void)out_size; (void)ws_size;
  const float* x     = (const float*)d_in[0];
  const float* nb    = (const float*)d_in[1];
  const float* pe    = (const float*)d_in[2];
  const float* ew    = (const float*)d_in[3];
  const float* ebv   = (const float*)d_in[4];
  const float* ainw  = (const float*)d_in[5];
  const float* ainb  = (const float*)d_in[6];
  const float* aoutw = (const float*)d_in[7];
  const float* aoutb = (const float*)d_in[8];
  const float* f1w   = (const float*)d_in[9];
  const float* f1b   = (const float*)d_in[10];
  const float* f2w   = (const float*)d_in[11];
  const float* f2b   = (const float*)d_in[12];
  const float* l1g   = (const float*)d_in[13];
  const float* l1b   = (const float*)d_in[14];
  const float* l2g   = (const float*)d_in[15];
  const float* l2b   = (const float*)d_in[16];
  const float* sinw  = (const float*)d_in[17];
  const float* sinb  = (const float*)d_in[18];
  const float* soutw = (const float*)d_in[19];
  const float* soutb = (const float*)d_in[20];
  const float* g1w   = (const float*)d_in[21];
  const float* g1b   = (const float*)d_in[22];
  const float* g2w   = (const float*)d_in[23];
  const float* g2b   = (const float*)d_in[24];
  const float* t1w   = (const float*)d_in[25];
  const float* t1b   = (const float*)d_in[26];
  const float* t2w   = (const float*)d_in[27];
  const float* t2b   = (const float*)d_in[28];
  const float* pw    = (const float*)d_in[29];
  const float* pb    = (const float*)d_in[30];

  _Float16* wf16 = (_Float16*)d_ws;
  float*    H32  = (float*)((char*)d_ws + WS_H32);
  _Float16* NH16 = (_Float16*)((char*)d_ws + WS_NH16);
  float*    HF   = (float*)((char*)d_ws + WS_HF);
  float*    out  = (float*)d_out;

  prep_kernel<<<(WF16_TOT + 255) / 256, 256, 0, stream>>>(
      ainw, aoutw, f1w, f2w, sinw, g1w, t1w, t2w, wf16);
  // 139264 sequences total, 32 per block (8 waves x 4 sequences)
  enc_kernel<<<4352, 256, ENC_LDS_TOT, stream>>>(
      x, nb, pe, ew, ebv, ainb, aoutb, f1b, f2b, l1g, l1b, l2g, l2b, wf16, H32, NH16, HF);
  soc_kernel<<<2048, 128, 4 * SOC_LDSW, stream>>>(
      wf16, H32, NH16, sinw, sinb, soutw, soutb, HF, out);
  heads_kernel<<<128, 128, 4 * HEADS_LDSW, stream>>>(
      wf16, HF, g1b, g2w, g2b, t1w, t1b, t2b, pw, pb, out);
}
